// GridSample_42623255445655
// MI455X (gfx1250) — matrix operand order repaired and checked by reading, not compiled, not run on hardware
//
#include <hip/hip_runtime.h>

// Fused BEV deformable-attention kernel for MI455X (gfx1250, wave32).
// Each wave owns 16 points; per-point state stays in LDS across all 3
// iterations; linears run on v_wmma_f32_16x16x32_f16; image gathers hit a
// channels-last copy in d_ws (coalesced 256B taps); all WMMA B-fragments
// are pre-packed to f16 in lane order so the hot loop loads them with one
// 32B vector load per lane.

typedef __attribute__((ext_vector_type(16))) _Float16 v16h;
typedef __attribute__((ext_vector_type(8)))  float    v8f;

#define XDIM 128
#define YDIM 128
#define ZDIM 8
#define EDIM 64
#define KPTS 3
#define NITER 3
#define FHH 96
#define FWW 320
#define NPIX (FHH * FWW)
#define PTS_PER_WAVE 16
#define WAVES_PER_BLOCK 4
#define THREADS_PER_BLOCK (WAVES_PER_BLOCK * 32)
#define PTS_PER_BLOCK (PTS_PER_WAVE * WAVES_PER_BLOCK)
#define NPTS (XDIM * YDIM * ZDIM)

// Packed-fragment arena: 10 fragments/iteration (2 offw + 8 W_l), 512 f16 each.
#define FRAGS_PER_ITER 10
#define FRAG_HALFS 512                       // 32 lanes x 16 halves
#define NFRAGS (NITER * FRAGS_PER_ITER)
#define IMG_BYTES ((size_t)NPIX * EDIM * sizeof(float))
#define WPK_BYTES ((size_t)NFRAGS * FRAG_HALFS * sizeof(_Float16))

// Same-wave LDS ordering fence: DS ops are in-order (DScnt); the "memory"
// clobber stops compiler reordering, s_wait_dscnt 0 drains outstanding DS ops.
#define LDS_FENCE() asm volatile("s_wait_dscnt 0x0" ::: "memory")

// A-matrix 16x32 f16 fragment: lane l holds row M=l&15;
// halves h: lanes 0-15 -> K = h; lanes 16-31 -> K = 16 + h.
__device__ __forceinline__ v16h load_A_frag(const float* row, int lane, int kfrag) {
  const int hi = lane >> 4;
  v16h a;
#pragma unroll
  for (int j = 0; j < 8; ++j) {
    const int k = 2 * j + hi * 16 + kfrag * 32;
    const float2 v = *reinterpret_cast<const float2*>(row + k);
    a[2 * j]     = (_Float16)v.x;
    a[2 * j + 1] = (_Float16)v.y;
  }
  return a;
}

// B-matrix 32x16 f16 fragment: lane l holds column N=l&15;
// lanes 0-15 cover K=kfrag*32+0..15, lanes 16-31 cover +16 (2 K per VGPR).
__device__ __forceinline__ v16h load_B_wl(const float* __restrict__ Wl,
                                          int lane, int ncol0, int kfrag) {
  const int n  = (lane & 15) + ncol0;
  const int kb = kfrag * 32 + (lane >> 4) * 16;
  v16h b;
#pragma unroll
  for (int h = 0; h < 16; ++h) b[h] = (_Float16)Wl[(kb + h) * EDIM + n];
  return b;
}

// Fused [W_off (6 cols) | W_w (3 cols) | zeros] B fragment.
__device__ __forceinline__ v16h load_B_offw(const float* __restrict__ Woff,
                                            const float* __restrict__ Ww,
                                            int lane, int kfrag) {
  const int n  = lane & 15;
  const int kb = kfrag * 32 + (lane >> 4) * 16;
  v16h b;
#pragma unroll
  for (int h = 0; h < 16; ++h) {
    const int k = kb + h;
    float v = 0.f;
    if (n < 6)      v = Woff[k * 6 + n];
    else if (n < 9) v = Ww[k * 3 + (n - 6)];
    b[h] = (_Float16)v;
  }
  return b;
}

__device__ __forceinline__ int clampi(int v, int lo, int hi) {
  return v < lo ? lo : (v > hi ? hi : v);
}

// (C,H,W) -> (H,W,C) channels-last transpose of img_feat into workspace.
__global__ __launch_bounds__(256) void transpose_img_kernel(
    const float* __restrict__ img, float* __restrict__ timg) {
  __shared__ float tile[64 * 65];
  const int t    = threadIdx.x;
  const int pix0 = blockIdx.x * 64;
#pragma unroll
  for (int j = 0; j < 16; ++j) {
    const int lin = t + j * 256;
    const int e   = lin >> 6;
    const int pp  = lin & 63;
    tile[pp * 65 + e] = img[(size_t)e * NPIX + pix0 + pp];
  }
  __syncthreads();
#pragma unroll
  for (int j = 0; j < 16; ++j) {
    const int lin = t + j * 256;
    const int pp  = lin >> 6;
    const int e   = lin & 63;
    timg[((size_t)(pix0 + pp)) * EDIM + e] = tile[pp * 65 + e];
  }
}

// Pre-pack all WMMA B-fragments (f16, lane-register order) into d_ws.
// Fragment layout per iteration: [offw f0, offw f1, Wl(t0,f0), Wl(t0,f1), ...].
__global__ __launch_bounds__(256) void pack_weights_kernel(
    const float* __restrict__ W_off, const float* __restrict__ W_w,
    const float* __restrict__ W_l, _Float16* __restrict__ wpk) {
  const int id = blockIdx.x * blockDim.x + threadIdx.x;
  if (id >= NFRAGS * 32) return;
  const int lane = id & 31;
  const int frag = id >> 5;
  const int it   = frag / FRAGS_PER_ITER;
  const int rem  = frag % FRAGS_PER_ITER;
  v16h b;
  if (rem < 2) {
    b = load_B_offw(W_off + it * EDIM * 6, W_w + it * EDIM * 3, lane, rem);
  } else {
    const int t = (rem - 2) >> 1;
    const int f = (rem - 2) & 1;
    b = load_B_wl(W_l + it * EDIM * EDIM, lane, t * 16, f);
  }
  *reinterpret_cast<v16h*>(wpk + (size_t)frag * FRAG_HALFS + lane * 16) = b;
}

template <bool FAST>
__global__ __launch_bounds__(THREADS_PER_BLOCK) void bev_deform_kernel(
    const float* __restrict__ Tv,    // (3,4) velo->cam
    const float* __restrict__ intr,  // (3,3)
    const float* __restrict__ img,   // FAST ? (H*W,64) : (64,H,W)
    const _Float16* __restrict__ wpk,// packed B fragments (FAST only)
    const float* __restrict__ bev0,  // (NPTS,64)
    const float* __restrict__ W_off, const float* __restrict__ b_off,
    const float* __restrict__ s_off,
    const float* __restrict__ W_w,   const float* __restrict__ b_w,
    const float* __restrict__ W_l,   const float* __restrict__ b_l,
    const float* __restrict__ ln_g,  const float* __restrict__ ln_b,
    float* __restrict__ out)         // (64,128,128)
{
  __shared__ float s_bev[WAVES_PER_BLOCK][PTS_PER_WAVE * EDIM]; // 16KB
  __shared__ float s_h  [WAVES_PER_BLOCK][PTS_PER_WAVE * EDIM]; // 16KB
  __shared__ float s_st [WAVES_PER_BLOCK][PTS_PER_WAVE * 16];   // 4KB
  __shared__ float s_pd [WAVES_PER_BLOCK][PTS_PER_WAVE * 12];   // 3KB

  const int tid   = threadIdx.x;
  const int wave  = tid >> 5;
  const int lane  = tid & 31;
  const int pbase = blockIdx.x * PTS_PER_BLOCK + wave * PTS_PER_WAVE;

  float* slab = s_bev[wave];
  float* hsl  = s_h[wave];
  float* stl  = s_st[wave];
  float* pd   = s_pd[wave];

  // Load this wave's 16x64 bev tile (4KB, coalesced b128s). Wave-private.
  {
    const float4* src = reinterpret_cast<const float4*>(bev0 + (size_t)pbase * EDIM);
    float4* dst = reinterpret_cast<float4*>(slab);
#pragma unroll
    for (int j = 0; j < 8; ++j) dst[lane + j * 32] = src[lane + j * 32];
  }

  // T = intrinsic @ T_velo2cam (tiny; per-thread).
  float Tm[12];
#pragma unroll
  for (int i = 0; i < 3; ++i)
#pragma unroll
    for (int j = 0; j < 4; ++j)
      Tm[i * 4 + j] = intr[i * 3 + 0] * Tv[0 * 4 + j] +
                      intr[i * 3 + 1] * Tv[1 * 4 + j] +
                      intr[i * 3 + 2] * Tv[2 * 4 + j];

  const int col = lane & 15;   // WMMA C/D column owned by this lane
  const int hi  = lane >> 4;

  LDS_FENCE();

  for (int it = 0; it < NITER; ++it) {
    const _Float16* wit = wpk + (size_t)it * FRAGS_PER_ITER * FRAG_HALFS;

    // ---- stage 1: fused offsets+logits GEMM: (16x64)@(64x16pad) ----
    {
      const float* arow = slab + col * EDIM;
      v16h a0 = load_A_frag(arow, lane, 0);
      v16h a1 = load_A_frag(arow, lane, 1);
      v16h bf0, bf1;
      if (FAST) {
        bf0 = *reinterpret_cast<const v16h*>(wit + lane * 16);
        bf1 = *reinterpret_cast<const v16h*>(wit + FRAG_HALFS + lane * 16);
      } else {
        bf0 = load_B_offw(W_off + it * EDIM * 6, W_w + it * EDIM * 3, lane, 0);
        bf1 = load_B_offw(W_off + it * EDIM * 6, W_w + it * EDIM * 3, lane, 1);
      }
      v8f c = {};
      c = __builtin_amdgcn_wmma_f32_16x16x32_f16(false, a0, false, bf0,
                                                 (short)0, c, false, false);
      c = __builtin_amdgcn_wmma_f32_16x16x32_f16(false, a1, false, bf1,
                                                 (short)0, c, false, false);
#pragma unroll
      for (int r = 0; r < 8; ++r) stl[(r + hi * 8) * 16 + col] = c[r];
    }
    LDS_FENCE();

    // ---- stage 2: per-point softmax + deformed sample coords (lanes 0-15) ----
    if (lane < 16) {
      const int m  = lane;
      const int p  = pbase + m;
      const int xg = p >> 10;          // / (Y*Z)
      const int yg = (p >> 3) & 127;
      const int zg = p & 7;
      const float px = xg * 0.8f;
      const float py = 51.2f - yg * 0.8f;
      const float pz = zg * 0.5f - 2.5f;
      const float q0 = Tm[0] * px + Tm[1] * py + Tm[2]  * pz + Tm[3];
      const float q1 = Tm[4] * px + Tm[5] * py + Tm[6]  * pz + Tm[7];
      const float q2 = Tm[8] * px + Tm[9] * py + Tm[10] * pz + Tm[11];
      const float cx = (q0 / q2) * (1.f / 640.f) - 1.f;
      const float cy = (q1 / q2) * (1.f / 192.f) - 1.f;
      const float sc = s_off[it];
      float lg[KPTS], ex[KPTS];
      float mx = -1e30f;
#pragma unroll
      for (int k = 0; k < KPTS; ++k) {
        lg[k] = stl[m * 16 + 6 + k] + b_w[it * KPTS + k];
        mx = fmaxf(mx, lg[k]);
      }
      float se = 0.f;
#pragma unroll
      for (int k = 0; k < KPTS; ++k) { ex[k] = __expf(lg[k] - mx); se += ex[k]; }
      const float inv = 1.f / se;
#pragma unroll
      for (int k = 0; k < KPTS; ++k) {
        const float ox = (stl[m * 16 + 2 * k]     + b_off[it * 6 + 2 * k])     * sc;
        const float oy = (stl[m * 16 + 2 * k + 1] + b_off[it * 6 + 2 * k + 1]) * sc;
        const float gx = cx + ox, gy = cy + oy;
        pd[m * 12 + 2 * k]     = ((gx + 1.f) * (float)FWW - 1.f) * 0.5f;
        pd[m * 12 + 2 * k + 1] = ((gy + 1.f) * (float)FHH - 1.f) * 0.5f;
        pd[m * 12 + 6 + k]     = ex[k] * inv;
      }
    }
    LDS_FENCE();

    // ---- stage 3: bilinear gather, weighted sum over K, bev += feat ----
    {
      const int e0 = lane * 2;  // two adjacent channels per lane
      for (int m = 0; m < PTS_PER_WAVE; ++m) {
        float acc0 = 0.f, acc1 = 0.f;
#pragma unroll
        for (int k = 0; k < KPTS; ++k) {
          const float fx = pd[m * 12 + 2 * k];
          const float fy = pd[m * 12 + 2 * k + 1];
          const float wk = pd[m * 12 + 6 + k];
          const float fx0 = floorf(fx), fy0 = floorf(fy);
          const int ix0 = (int)fx0, iy0 = (int)fy0;
          const float wx1 = fx - fx0, wx0 = 1.f - wx1;
          const float wy1 = fy - fy0, wy0 = 1.f - wy1;
#pragma unroll
          for (int t = 0; t < 4; ++t) {
            const int ix = ix0 + (t & 1);
            const int iy = iy0 + (t >> 1);
            const bool ok = (ix >= 0) & (ix < FWW) & (iy >= 0) & (iy < FHH);
            const int ixc = clampi(ix, 0, FWW - 1);
            const int iyc = clampi(iy, 0, FHH - 1);
            float v0, v1;
            if (FAST) {
              const float2 v = *reinterpret_cast<const float2*>(
                  img + (size_t)(iyc * FWW + ixc) * EDIM + e0);
              v0 = v.x; v1 = v.y;
            } else {
              const size_t sp = (size_t)iyc * FWW + ixc;
              v0 = img[(size_t)e0 * NPIX + sp];
              v1 = img[(size_t)(e0 + 1) * NPIX + sp];
            }
            const float wt = ok ? wk * ((t & 1) ? wx1 : wx0) *
                                       ((t >> 1) ? wy1 : wy0)
                                : 0.f;
            acc0 += wt * v0;
            acc1 += wt * v1;
          }
        }
        float2* bp = reinterpret_cast<float2*>(slab + m * EDIM + e0);
        float2 b = *bp;
        b.x += acc0; b.y += acc1;
        *bp = b;
      }
    }
    LDS_FENCE();

    // ---- stage 4: h = bev @ W_l + b_l  (16x64 @ 64x64, 8 WMMAs) ----
    {
      const float* arow = slab + col * EDIM;
      v16h a0 = load_A_frag(arow, lane, 0);
      v16h a1 = load_A_frag(arow, lane, 1);
#pragma unroll
      for (int t = 0; t < 4; ++t) {
        v16h bf0, bf1;
        if (FAST) {
          const _Float16* wt_ = wit + (size_t)(2 + t * 2) * FRAG_HALFS;
          bf0 = *reinterpret_cast<const v16h*>(wt_ + lane * 16);
          bf1 = *reinterpret_cast<const v16h*>(wt_ + FRAG_HALFS + lane * 16);
        } else {
          bf0 = load_B_wl(W_l + it * EDIM * EDIM, lane, t * 16, 0);
          bf1 = load_B_wl(W_l + it * EDIM * EDIM, lane, t * 16, 1);
        }
        v8f c = {};
        c = __builtin_amdgcn_wmma_f32_16x16x32_f16(false, a0, false, bf0,
                                                   (short)0, c, false, false);
        c = __builtin_amdgcn_wmma_f32_16x16x32_f16(false, a1, false, bf1,
                                                   (short)0, c, false, false);
        const int n  = t * 16 + col;
        const float bl = b_l[it * EDIM + n];
#pragma unroll
        for (int r = 0; r < 8; ++r) hsl[(r + hi * 8) * EDIM + n] = c[r] + bl;
      }
    }
    LDS_FENCE();

    // ---- stage 5: LayerNorm over E, residual add into bev ----
    {
      const int m  = col;        // point handled by this lane
      const int cb = hi * 32;    // channel half
      float s = 0.f, s2 = 0.f;
#pragma unroll
      for (int c = 0; c < 32; ++c) {
        const float v = hsl[m * EDIM + cb + c];
        s += v; s2 += v * v;
      }
      s  += __shfl_xor(s, 16, 32);
      s2 += __shfl_xor(s2, 16, 32);
      const float mu  = s * (1.f / 64.f);
      const float var = s2 * (1.f / 64.f) - mu * mu;
      const float rs  = rsqrtf(var + 1e-5f);
#pragma unroll
      for (int c = 0; c < 32; ++c) {
        const int e = cb + c;
        const float v  = hsl[m * EDIM + e];
        const float hn = (v - mu) * rs * ln_g[it * EDIM + e] + ln_b[it * EDIM + e];
        slab[m * EDIM + e] += hn;
      }
    }
    LDS_FENCE();
  }

  // ---- output: out[e,y,x] = mean_z bev[x,y,z,e]; 16 pts = 2 (x,y) groups ----
  {
    const int xybase = pbase >> 3;
#pragma unroll
    for (int g = 0; g < 2; ++g) {
      const int xy = xybase + g;
      const int xg = xy >> 7;
      const int yg = xy & 127;
#pragma unroll
      for (int c = 0; c < 2; ++c) {
        const int e = lane + 32 * c;
        float s = 0.f;
#pragma unroll
        for (int z = 0; z < 8; ++z) s += slab[(g * 8 + z) * EDIM + e];
        out[((size_t)e * YDIM + yg) * XDIM + xg] = s * 0.125f;
      }
    }
  }
}

extern "C" void kernel_launch(void* const* d_in, const int* in_sizes, int n_in,
                              void* d_out, int out_size, void* d_ws, size_t ws_size,
                              hipStream_t stream) {
  const float* Tv    = (const float*)d_in[0];
  const float* intr  = (const float*)d_in[1];
  const float* img   = (const float*)d_in[2];
  const float* bev   = (const float*)d_in[3];
  const float* W_off = (const float*)d_in[4];
  const float* b_off = (const float*)d_in[5];
  const float* s_off = (const float*)d_in[6];
  const float* W_w   = (const float*)d_in[7];
  const float* b_w   = (const float*)d_in[8];
  const float* W_l   = (const float*)d_in[9];
  const float* b_l   = (const float*)d_in[10];
  const float* ln_g  = (const float*)d_in[11];
  const float* ln_b  = (const float*)d_in[12];
  float* out = (float*)d_out;

  const int nblocks = NPTS / PTS_PER_BLOCK;

  if (ws_size >= IMG_BYTES + WPK_BYTES) {
    float*     timg = (float*)d_ws;
    _Float16*  wpk  = (_Float16*)((char*)d_ws + IMG_BYTES);
    transpose_img_kernel<<<NPIX / 64, 256, 0, stream>>>(img, timg);
    pack_weights_kernel<<<(NFRAGS * 32 + 255) / 256, 256, 0, stream>>>(
        W_off, W_w, W_l, wpk);
    bev_deform_kernel<true><<<nblocks, THREADS_PER_BLOCK, 0, stream>>>(
        Tv, intr, timg, wpk, bev, W_off, b_off, s_off, W_w, b_w, W_l, b_l,
        ln_g, ln_b, out);
  } else {
    bev_deform_kernel<false><<<nblocks, THREADS_PER_BLOCK, 0, stream>>>(
        Tv, intr, img, (const _Float16*)nullptr, bev, W_off, b_off, s_off,
        W_w, b_w, W_l, b_l, ln_g, ln_b, out);
  }
}
